// BOREP_41162966565492
// MI455X (gfx1250) — matrix operand order repaired and checked by reading, not compiled, not run on hardware
//
#include <hip/hip_runtime.h>
#include <stdint.h>

// out[s,b,o] = sum_i x[s,b,i]*W[o,i] + bias[o]
// M=16384, K=1024, N=4096, fp32 in/out.
//
// Pass 1: fp32 -> bf16 (RNE) for X and W into d_ws (bandwidth bound, ~5us).
// Pass 2: bf16 WMMA GEMM. Operands (42 MB bf16) are L2-resident (192 MB L2).
//         Staging uses GLOBAL_LOAD_ASYNC_TO_LDS_B128 (ASYNCcnt DMA, no VGPR
//         round-trip, no VALU in the hot loop), double-buffered LDS,
//         v_wmma_f32_16x16x32_bf16 with f32 accumulation.

typedef __attribute__((ext_vector_type(16))) __bf16        v16bf;
typedef __attribute__((ext_vector_type(8)))  float         v8f;
typedef __attribute__((ext_vector_type(4)))  float         v4f;
typedef __attribute__((ext_vector_type(4)))  unsigned int  v4u;

#define M_TOTAL 16384
#define K_DIM   1024
#define N_DIM   4096
#define BM      128
#define BN      128
#define BK      32
#define NK      (K_DIM / BK)          // 32 K-steps
#define ROW_BYTES  80                 // 64B bf16 row + 16B pad (bank-conflict free)
#define TILE_BYTES (BM * ROW_BYTES)   // 10240
#define BUF_BYTES  (2 * TILE_BYTES)   // A tile + B tile per buffer

#define XBF_BYTES  ((size_t)M_TOTAL * K_DIM * 2)   // 33,554,432
#define WBF_BYTES  ((size_t)N_DIM  * K_DIM * 2)    //  8,388,608

union Frag {
    v4u   q[2];   // two ds_load_b128 halves
    v16bf v;      // WMMA operand view
};

__device__ __forceinline__ unsigned int pk_bf16(float lo, float hi) {
    unsigned int ulo = __float_as_uint(lo);
    unsigned int uhi = __float_as_uint(hi);
    ulo += 0x7fffu + ((ulo >> 16) & 1u);
    uhi += 0x7fffu + ((uhi >> 16) & 1u);
    return (ulo >> 16) | (uhi & 0xffff0000u);
}

// GLOBAL_LOAD_ASYNC_TO_LDS_B128 (GVS mode): LDS[lds_addr] <= MEM[sbase + voff],
// tracked by ASYNCcnt. lds_addr = raw LDS byte address (low 32 bits of generic
// pointer to __shared__), voff = per-lane byte offset, sbase = uniform base.
__device__ __forceinline__ void async_ld_b128(unsigned lds_addr, unsigned voff,
                                              const void* sbase) {
    asm volatile("global_load_async_to_lds_b128 %0, %1, %2"
                 :: "v"(lds_addr), "v"(voff), "s"(sbase)
                 : "memory");
}

__device__ __forceinline__ void wait_asynccnt0() {
    asm volatile("s_wait_asynccnt 0x0" ::: "memory");
}

// ---------------- Pass 1: fp32 -> bf16 conversion (8 elems/thread) ----------
__global__ __launch_bounds__(256)
void cvt_f32_to_bf16(const float* __restrict__ src, v4u* __restrict__ dst, int n8) {
    int i = blockIdx.x * blockDim.x + threadIdx.x;
    if (i >= n8) return;
    v4f a = ((const v4f*)src)[2 * i];
    v4f b = ((const v4f*)src)[2 * i + 1];
    v4u o;
    o.x = pk_bf16(a.x, a.y);
    o.y = pk_bf16(a.z, a.w);
    o.z = pk_bf16(b.x, b.y);
    o.w = pk_bf16(b.z, b.w);
    dst[i] = o;
}

// ---------------- Pass 2: bf16 WMMA GEMM ------------------------------------
__global__ __launch_bounds__(256)
void borep_wmma_bf16(const void* __restrict__ Xbf,   // [16384,1024] bf16
                     const void* __restrict__ Wbf,   // [4096,1024]  bf16
                     const float* __restrict__ bias,
                     float* __restrict__ out) {
    __shared__ __align__(16) char lds[2 * BUF_BYTES];
    const unsigned lds0 = (unsigned)(uintptr_t)(void*)&lds[0];  // raw LDS byte addr

    const int tid = threadIdx.x;
    const int bm0 = blockIdx.y * BM;
    const int bn0 = blockIdx.x * BN;

    // ---- staging mapping: 256 threads x 32B cover a 128x32 bf16 tile -------
    const int crow = tid >> 1;              // 0..127
    const int ckb  = (tid & 1) << 5;        // byte 0 or 32 within 64B row
    const unsigned aGlob = (unsigned)((bm0 + crow) * (K_DIM * 2) + ckb);
    const unsigned bGlob = (unsigned)((bn0 + crow) * (K_DIM * 2) + ckb);
    const unsigned ldsRow = (unsigned)(crow * ROW_BYTES + ckb);

    // ---- compute mapping: 8 waves in 2(M) x 4(N); wave tile = 64x32 --------
    const int wave  = tid >> 5;
    const int lane  = tid & 31;
    const int waveM = wave >> 2;            // 0..1
    const int waveN = wave & 3;             // 0..3
    const int l16   = lane & 15;
    const int hi    = lane >> 4;
    const int aoff  = hi << 4;              // A frag: K 0-7 vs 8-15   (byte 0/16)
    const int boff  = hi << 4;              // B frag: K 0-7 vs 8-15   (byte 0/16)

    v8f acc[4][2] = {};

    // ---- prologue: async-stage K-tile 0 into buffer 0 ----------------------
    {
        const unsigned la = lds0 + ldsRow;               // A tile
        const unsigned lb = la + TILE_BYTES;             // B tile
        async_ld_b128(la,      aGlob,      Xbf);
        async_ld_b128(la + 16, aGlob + 16, Xbf);
        async_ld_b128(lb,      bGlob,      Wbf);
        async_ld_b128(lb + 16, bGlob + 16, Wbf);
    }
    wait_asynccnt0();
    __syncthreads();

    // ---- main K loop, double-buffered DMA pipeline -------------------------
    for (int kt = 0; kt < NK; ++kt) {
        const int  cur  = kt & 1;
        const int  nxt  = cur ^ 1;
        const bool more = (kt + 1) < NK;

        if (more) {   // DMA next K-tile into the other buffer while computing
            const unsigned kb = (unsigned)((kt + 1) * (BK * 2));
            const unsigned la = lds0 + (unsigned)nxt * BUF_BYTES + ldsRow;
            const unsigned lb = la + TILE_BYTES;
            async_ld_b128(la,      aGlob + kb,      Xbf);
            async_ld_b128(la + 16, aGlob + kb + 16, Xbf);
            async_ld_b128(lb,      bGlob + kb,      Wbf);
            async_ld_b128(lb + 16, bGlob + kb + 16, Wbf);
            if (kt + 2 < NK) {
                __builtin_prefetch((const char*)Xbf + aGlob + kb + BK * 2, 0, 0);
                __builtin_prefetch((const char*)Wbf + bGlob + kb + BK * 2, 0, 0);
            }
        }

        // compute on current buffer: 8 x v_wmma_f32_16x16x32_bf16 per wave
        {
            const char* Ab = lds + cur * BUF_BYTES;
            const char* Bb = Ab + TILE_BYTES;

            Frag bfr[2];
#pragma unroll
            for (int j = 0; j < 2; ++j) {
                const char* p = Bb + (waveN * 32 + j * 16 + l16) * ROW_BYTES + boff;
                bfr[j].q[0] = *(const v4u*)(p);
                bfr[j].q[1] = *(const v4u*)(p + 32);
            }
#pragma unroll
            for (int i = 0; i < 4; ++i) {
                Frag afr;
                const char* p = Ab + (waveM * 64 + i * 16 + l16) * ROW_BYTES + aoff;
                afr.q[0] = *(const v4u*)(p);
                afr.q[1] = *(const v4u*)(p + 32);
#pragma unroll
                for (int j = 0; j < 2; ++j) {
                    acc[i][j] = __builtin_amdgcn_wmma_f32_16x16x32_bf16(
                        false, afr.v, false, bfr[j].v,
                        (short)0, acc[i][j], false, false);
                }
            }
        }

        if (more) wait_asynccnt0();   // own DMA landed; barrier makes it global
        __syncthreads();
    }

    // ---- epilogue: bias + coalesced stores (C/D: reg r -> row, lane -> col) -
#pragma unroll
    for (int j = 0; j < 2; ++j) {
        const int   gn = bn0 + waveN * 32 + j * 16 + l16;
        const float bv = bias[gn];
#pragma unroll
        for (int i = 0; i < 4; ++i) {
            const int gmb = bm0 + waveM * 64 + i * 16 + hi * 8;
#pragma unroll
            for (int r = 0; r < 8; ++r) {
                out[(size_t)(gmb + r) * N_DIM + gn] = acc[i][j][r] + bv;
            }
        }
    }
}

extern "C" void kernel_launch(void* const* d_in, const int* in_sizes, int n_in,
                              void* d_out, int out_size, void* d_ws, size_t ws_size,
                              hipStream_t stream) {
    const float* X    = (const float*)d_in[0];   // [256,64,1024]
    const float* W    = (const float*)d_in[1];   // [4096,1024]
    const float* bias = (const float*)d_in[2];   // [4096]
    float*       out  = (float*)d_out;           // [16384,4096]

    void* Xbf = d_ws;                                   // 33.5 MB bf16
    void* Wbf = (char*)d_ws + XBF_BYTES;                //  8.4 MB bf16

    // Pass 1: fp32 -> bf16 (8 elems per thread, exact division)
    {
        const int n8x = M_TOTAL * K_DIM / 8;            // 2,097,152
        const int n8w = N_DIM * K_DIM / 8;              //   524,288
        hipLaunchKernelGGL(cvt_f32_to_bf16, dim3(n8x / 256), dim3(256), 0, stream,
                           X, (v4u*)Xbf, n8x);
        hipLaunchKernelGGL(cvt_f32_to_bf16, dim3(n8w / 256), dim3(256), 0, stream,
                           W, (v4u*)Wbf, n8w);
    }

    // Pass 2: WMMA GEMM
    dim3 grid(N_DIM / BN, M_TOTAL / BM);                // (32, 128)
    hipLaunchKernelGGL(borep_wmma_bf16, grid, dim3(256), 0, stream,
                       (const void*)Xbf, (const void*)Wbf, bias, out);
}
